// ProxyAttentionBlock_40020505264419
// MI455X (gfx1250) — matrix operand order repaired and checked by reading, not compiled, not run on hardware
//
#include <hip/hip_runtime.h>
#include <hip/hip_bf16.h>
#include <stdint.h>

#ifndef __has_builtin
#define __has_builtin(x) 0
#endif

// ---------------------------------------------------------------------------
// CDNA5 (gfx1250) fused proxy-attention block.
// v_wmma_f32_16x16x32_bf16 for all heavy GEMMs; weights streamed to LDS with
// the Tensor Data Mover (double buffered, TENSORcnt-synchronized, LDS-padded
// to a bank-conflict-free stride).
// ---------------------------------------------------------------------------

typedef float v8f  __attribute__((ext_vector_type(8)));
typedef __bf16 v16bf __attribute__((ext_vector_type(16)));

union Frag32B {               // 32-byte WMMA operand fragment (16 x bf16)
    uint4  q[2];
    __bf16 h[16];
    v16bf  v;
};

static __device__ __forceinline__ __bf16 f2bf(float f) { return (__bf16)f; }

static __device__ __forceinline__ v8f wmma_bf16(v16bf a, v16bf b, v8f c) {
    return __builtin_amdgcn_wmma_f32_16x16x32_bf16(
        false, a, false, b, (short)0, c, false, false);
}

#define C 512
#define HEADS 8
#define DH 64
#define AGENT 16
#define N_TOK 4096
#define B_SZ 32
#define ROWS 64          // rows per workgroup
#define KSTEPS 16        // 512 / 32

#define ASTRIDE 520      // tileA row stride (bf16): 4-dword bank skew
#define WSTRIDE 40       // wslab per-col stride (bf16): 20-dword bank skew

// ------------------------- workspace layout (bytes) -------------------------
#define WS_WQT     0u                         // 512*512 bf16  (WqT[n][k])
#define WS_WPT     524288u                    // 512*512 bf16  (WpT[n][k])
#define WS_KT      1048576u                   // [8][16][64] bf16 (scaled k)
#define WS_VPROJ   1064960u                   // [16][512] f32
#define WS_GSUM    1097728u                   // [32][16] f32
#define WS_GMAX    1099776u                   // [32][16] u32
#define WS_VPRIME  1101824u                   // [32][8][64][32] bf16
#define WS_ATTN    2150400u                   // [32][8][4096][16] bf16

// ---------------------------------------------------------------------------
// Tensor Data Mover: async copy of a 512x32 bf16 slab (row stride 512 elems)
// from global into LDS laid out [512][WSTRIDE] (TDM pads 16B after every 64B).
// Falls back to a cooperative synchronous copy when the builtin is missing.
// ---------------------------------------------------------------------------
#if __has_builtin(__builtin_amdgcn_tensor_load_to_lds)
#define USE_TDM 1
typedef unsigned int u32x4 __attribute__((ext_vector_type(4)));
typedef int i32x4 __attribute__((ext_vector_type(4)));
typedef int i32x8 __attribute__((ext_vector_type(8)));

static __device__ __forceinline__ void tdm_load_slab(const __bf16* g, unsigned lds_off) {
    unsigned long long ga = (unsigned long long)(uintptr_t)g;
    u32x4 g0;
    g0[0] = 1u;                                              // count=1, user D#
    g0[1] = lds_off;                                         // lds_addr
    g0[2] = (unsigned)ga;                                    // global_addr lo
    g0[3] = (unsigned)((ga >> 32) & 0x1FFFFFFull) | (2u << 30); // addr hi | type=2
    i32x8 g1;
    // data_size=2B | pad_enable | pad_interval=3 (16 dw = 64B) | pad_amount=3 (4 dw = 16B)
    g1[0] = (int)((1u << 16) | (1u << 20) | (3u << 22) | (3u << 25));
    g1[1] = (int)((512u & 0xFFFFu) << 16);                   // tensor_dim0 lo
    g1[2] = (int)(((512u >> 16) & 0xFFFFu) | ((512u & 0xFFFFu) << 16)); // d0 hi | d1 lo
    g1[3] = (int)(((512u >> 16) & 0xFFFFu) | (32u << 16));   // d1 hi | tile_dim0=32
    g1[4] = (int)512;                                        // tile_dim1=512
    g1[5] = (int)512;                                        // tensor_dim0_stride=512
    g1[6] = 0;
    g1[7] = 0;
    i32x4 g2 = {0, 0, 0, 0};
    i32x4 g3 = {0, 0, 0, 0};
#if __clang_major__ >= 23
    i32x8 g4 = {0, 0, 0, 0, 0, 0, 0, 0};
    __builtin_amdgcn_tensor_load_to_lds(g0, g1, g2, g3, g4, 0);
#else
    __builtin_amdgcn_tensor_load_to_lds(g0, g1, g2, g3, 0);
#endif
}
#else
#define USE_TDM 0
#endif

static __device__ __forceinline__ void slab_issue(__bf16* dstLds, const __bf16* srcG,
                                                  int wave, int tid) {
#if USE_TDM
    if (wave == 0) tdm_load_slab(srcG, (unsigned)(uintptr_t)dstLds);
    (void)tid;
#else
    // cooperative copy: 512 cols x 32 K, uint4 (8 bf16) granules, padded stride
    for (int i = 0; i < 8; ++i) {
        const int idx = i * 256 + tid;        // 0..2047
        const int col = idx >> 2;
        const int seg = idx & 3;
        *(uint4*)(dstLds + col * WSTRIDE + seg * 8) =
            *(const uint4*)(srcG + (size_t)col * C + seg * 8);
    }
    (void)wave;
#endif
}

static __device__ __forceinline__ void slab_wait(bool more_in_flight, int wave) {
#if USE_TDM
    if (wave == 0) {
        if (more_in_flight) __builtin_amdgcn_s_wait_tensorcnt((short)1);
        else                __builtin_amdgcn_s_wait_tensorcnt((short)0);
    }
#else
    (void)more_in_flight; (void)wave;
#endif
}

static __device__ __forceinline__ Frag32B load_frag16(const __bf16* p) {
    Frag32B f;
    f.q[0] = *(const uint4*)(p);
    f.q[1] = *(const uint4*)(p + 8);
    return f;
}

// ---------------------------------------------------------------------------
// Kernel 1: transpose + convert Wq, Wp to bf16 (B operand K-contiguous).
// ---------------------------------------------------------------------------
__global__ void pk_prep_w(const float* __restrict__ Wq,
                          const float* __restrict__ Wp,
                          __bf16* __restrict__ WqT,
                          __bf16* __restrict__ WpT) {
    unsigned idx = blockIdx.x * blockDim.x + threadIdx.x;   // 0..524287
    if (idx < 262144u) {
        unsigned n = idx >> 9, k = idx & 511u;
        WqT[n * 512u + k] = f2bf(Wq[k * 512u + n]);
    } else {
        unsigned j = idx - 262144u;
        unsigned n = j >> 9, k = j & 511u;
        WpT[n * 512u + k] = f2bf(Wp[k * 512u + n]);
    }
}

// ---------------------------------------------------------------------------
// Kernel 2: tiny k/v projections; 1/sqrt(d)=0.125 folded into kT.
// ---------------------------------------------------------------------------
__global__ void pk_prep_kv(const float* __restrict__ agent_k,
                           const float* __restrict__ Wk,
                           const float* __restrict__ bk,
                           const float* __restrict__ agent_v,
                           const float* __restrict__ Wv,
                           const float* __restrict__ bv,
                           __bf16* __restrict__ kT,
                           float* __restrict__ vproj) {
    unsigned gid = blockIdx.x * blockDim.x + threadIdx.x;   // 0..16383
    unsigned a = (gid >> 9) & 15u;
    unsigned j = gid & 511u;
    if (gid < 8192u) {
        float acc = bk[j];
        for (int i = 0; i < 512; ++i) acc += agent_k[a * 512u + i] * Wk[i * 512u + j];
        unsigned h = j >> 6, kd = j & 63u;
        kT[(h * 16u + a) * 64u + kd] = f2bf(0.125f * acc);
    } else {
        float acc = bv[j];
        for (int i = 0; i < 512; ++i) acc += agent_v[a * 512u + i] * Wv[i * 512u + j];
        vproj[a * 512u + j] = acc;
    }
}

// ---------------------------------------------------------------------------
// Kernel 3 (pass 1): q GEMM (WMMA, TDM-staged weights) + per-head scores +
// softmax over 16 agents + stats + attn store (bf16).
// Wave tile: 32 rows x 128 cols (2 A-frags, 8 B-frags, 16 WMMAs / K-step).
// Grid: 2048 WGs (32 b x 64 row tiles), 256 threads (8 waves).
// ---------------------------------------------------------------------------
__global__ void __launch_bounds__(256)
pk_qattn(const float* __restrict__ x,
         const __bf16* __restrict__ WqT,
         const float* __restrict__ bq,
         const __bf16* __restrict__ kT,
         __bf16* __restrict__ attnA,
         float* __restrict__ gsum,
         unsigned* __restrict__ gmax) {
    __shared__ __bf16 tileA[ROWS * ASTRIDE];       // ~65 KB: x (bf16), later q
    __shared__ __bf16 wslab[2][C * WSTRIDE];       // 2 x 40 KB padded K-slabs
    __shared__ float    sSum[AGENT];
    __shared__ unsigned sMax[AGENT];

    const int tid   = threadIdx.x;
    const int wave  = tid >> 5;
    const int lane  = tid & 31;
    const int l16   = lane & 15;
    const int lhalf = lane >> 4;
    const int wm    = wave >> 2;          // 0..1 : 32-row block
    const int wn    = wave & 3;           // 0..3 : 128-col block

    const int b  = blockIdx.x >> 6;
    const int n0 = (blockIdx.x & 63) << 6;

    if (tid < AGENT) { sSum[tid] = 0.f; sMax[tid] = 0u; }

    // ---- stage x tile -> LDS bf16 (row-major, K contiguous, padded) ----
    const float* xbase = x + (size_t)(b * N_TOK + n0) * C;
    for (int i = 0; i < 32; ++i) {
        const int g = i * 256 + tid;          // group of 4 elems
        const int row = g >> 7;
        const int off = (g & 127) << 2;
        float4 f = *(const float4*)(xbase + (size_t)row * C + off);
        union { __bf16 h[4]; uint2 u; } pk;
        pk.h[0] = f2bf(f.x); pk.h[1] = f2bf(f.y);
        pk.h[2] = f2bf(f.z); pk.h[3] = f2bf(f.w);
        *(uint2*)&tileA[row * ASTRIDE + off] = pk.u;
    }

    slab_issue(&wslab[0][0], WqT, wave, tid);      // slab for k0 = 0
    slab_wait(false, wave);
    __syncthreads();

    // ---------------- GEMM: q[64x512] = x @ Wq -----------------
    v8f acc[16];
    #pragma unroll
    for (int t = 0; t < 16; ++t) acc[t] = (v8f){0,0,0,0,0,0,0,0};

    for (int ks = 0; ks < KSTEPS; ++ks) {
        const int buf = ks & 1;
        if (ks + 1 < KSTEPS)
            slab_issue(&wslab[buf ^ 1][0], WqT + (ks + 1) * 32, wave, tid);
        slab_wait(ks + 1 < KSTEPS, wave);
        __syncthreads();                      // current slab visible

        const __bf16* arow = tileA + ks * 32 + 16 * lhalf;
        Frag32B af0 = load_frag16(arow + (wm * 32 + l16) * ASTRIDE);
        Frag32B af1 = load_frag16(arow + (wm * 32 + 16 + l16) * ASTRIDE);
        // software-rotated B fragments: loads for t+1 overlap WMMA t
        Frag32B bcur = load_frag16(&wslab[buf][(wn * 128 + l16) * WSTRIDE + 16 * lhalf]);
        #pragma unroll
        for (int t = 0; t < 8; ++t) {
            Frag32B bnext;
            if (t < 7)
                bnext = load_frag16(&wslab[buf][(wn * 128 + (t + 1) * 16 + l16) * WSTRIDE + 16 * lhalf]);
            acc[t]     = wmma_bf16(af0.v, bcur.v, acc[t]);
            acc[8 + t] = wmma_bf16(af1.v, bcur.v, acc[8 + t]);
            bcur = bnext;
        }
        __syncthreads();                      // reads done before slab rewrite
    }

    // bias + q tile -> LDS (overwrite x tile; all reads completed above)
    #pragma unroll
    for (int mt = 0; mt < 2; ++mt) {
        #pragma unroll
        for (int t = 0; t < 8; ++t) {
            const int col = wn * 128 + t * 16 + l16;
            const float bqv = bq[col];
            #pragma unroll
            for (int e = 0; e < 8; ++e) {
                const int row = wm * 32 + mt * 16 + e + 8 * lhalf;
                tileA[row * ASTRIDE + col] = f2bf(acc[mt * 8 + t][e] + bqv);
            }
        }
    }
    __syncthreads();

    // ------------- scores + softmax: wave == head, 4 row tiles -------------
    const int h = wave;
    float lsum = 0.f, lmax = 0.f;
    for (int rt = 0; rt < 4; ++rt) {
        v8f sc = (v8f){0,0,0,0,0,0,0,0};
        #pragma unroll
        for (int kk = 0; kk < 64; kk += 32) {
            Frag32B af, bf;
            const __bf16* ap = tileA + (rt * 16 + l16) * ASTRIDE + h * DH + kk + 16 * lhalf;
            af.q[0] = *(const uint4*)(ap);
            af.q[1] = *(const uint4*)(ap + 8);
            bf = load_frag16(kT + (h * 16 + l16) * 64 + kk + 16 * lhalf);
            sc = wmma_bf16(af.v, bf.v, sc);
        }
        #pragma unroll
        for (int e = 0; e < 8; ++e) {
            float s = sc[e];                  // agent col = l16
            float m = s;
            m = fmaxf(m, __shfl_xor(m, 1, 16));
            m = fmaxf(m, __shfl_xor(m, 2, 16));
            m = fmaxf(m, __shfl_xor(m, 4, 16));
            m = fmaxf(m, __shfl_xor(m, 8, 16));
            float p = __expf(s - m);
            float sm = p;
            sm += __shfl_xor(sm, 1, 16);
            sm += __shfl_xor(sm, 2, 16);
            sm += __shfl_xor(sm, 4, 16);
            sm += __shfl_xor(sm, 8, 16);
            const float at = p / sm;
            lsum += at;
            lmax = fmaxf(lmax, at);
            const int rowg = n0 + rt * 16 + e + 8 * lhalf;
            attnA[((size_t)(b * HEADS + h) * N_TOK + rowg) * AGENT + l16] = f2bf(at);
        }
    }
    lsum += __shfl_xor(lsum, 16, 32);
    lmax = fmaxf(lmax, __shfl_xor(lmax, 16, 32));
    if (lhalf == 0) {
        atomicAdd(&sSum[l16], lsum);
        atomicMax(&sMax[l16], __float_as_uint(lmax));   // attn > 0 -> monotone
    }
    __syncthreads();
    if (tid < AGENT) {
        atomicAdd(&gsum[b * AGENT + tid], sSum[tid]);
        atomicMax(&gmax[b * AGENT + tid], sMax[tid]);
    }
}

// ---------------------------------------------------------------------------
// Kernel 4: SE gate -> gated, K-padded v' per batch (bf16, B-operand layout).
// ---------------------------------------------------------------------------
__global__ void pk_gate(const float* __restrict__ gsum,
                        const unsigned* __restrict__ gmax,
                        const float* __restrict__ ca_w1,
                        const float* __restrict__ ca_w2,
                        const float* __restrict__ vproj,
                        __bf16* __restrict__ vprimeT) {
    __shared__ float gateS[AGENT];
    const int b = blockIdx.x;
    if (threadIdx.x == 0) {
        float za = 0.f, zm = 0.f;
        for (int a = 0; a < AGENT; ++a) {
            const float avg = gsum[b * AGENT + a] * (1.0f / 32768.0f);
            const float mx  = __uint_as_float(gmax[b * AGENT + a]);
            za += avg * ca_w1[a];
            zm += mx * ca_w1[a];
        }
        za = fmaxf(za, 0.f);
        zm = fmaxf(zm, 0.f);
        const float t = za + zm;
        for (int a = 0; a < AGENT; ++a)
            gateS[a] = 1.0f / (1.0f + __expf(-(t * ca_w2[a])));
    }
    __syncthreads();
    for (int i = 0; i < 64; ++i) {
        const unsigned idx = threadIdx.x + i * 256u;      // 0..16383
        const unsigned kk = idx & 31u;
        const unsigned d  = (idx >> 5) & 63u;
        const unsigned h  = idx >> 11;
        const float v = (kk < AGENT) ? gateS[kk] * vproj[kk * 512u + h * 64u + d] : 0.f;
        vprimeT[(size_t)b * 16384u + idx] = f2bf(v);
    }
}

// ---------------------------------------------------------------------------
// Kernel 5 (pass 2): atte = attn @ v' (WMMA, K padded 16->32), then fused
// out = atte @ Wp + bp with TDM-staged Wp slabs.
// ---------------------------------------------------------------------------
__global__ void __launch_bounds__(256)
pk_avout(const __bf16* __restrict__ attnA,
         const __bf16* __restrict__ vprimeT,
         const __bf16* __restrict__ WpT,
         const float* __restrict__ bp,
         float* __restrict__ out) {
    __shared__ __bf16 tileA[ROWS * ASTRIDE];       // atte tile, bf16 (padded)
    __shared__ __bf16 wslab[2][C * WSTRIDE];

    const int tid   = threadIdx.x;
    const int wave  = tid >> 5;
    const int lane  = tid & 31;
    const int l16   = lane & 15;
    const int lhalf = lane >> 4;
    const int wm    = wave >> 2;
    const int wn    = wave & 3;

    const int b  = blockIdx.x >> 6;
    const int n0 = (blockIdx.x & 63) << 6;

    v8f acc[16];
    #pragma unroll
    for (int t = 0; t < 16; ++t) acc[t] = (v8f){0,0,0,0,0,0,0,0};

    slab_issue(&wslab[0][0], WpT, wave, tid);      // prefetch Wp slab 0 early

    // ---- atte: per wave rows {wm*32, +16}, cols wn*128 (heads wn*2..+1) ----
    #pragma unroll
    for (int mt = 0; mt < 2; ++mt) {
        #pragma unroll
        for (int hh = 0; hh < 2; ++hh) {
            const int h = wn * 2 + hh;
            Frag32B af;
            const __bf16* ap = attnA +
                ((size_t)(b * HEADS + h) * N_TOK + n0 + wm * 32 + mt * 16 + l16) * AGENT + 8 * lhalf;
            af.q[0] = *(const uint4*)ap;
            af.q[1] = make_uint4(0u, 0u, 0u, 0u);            // K pad 16..31 = 0
            #pragma unroll
            for (int dt = 0; dt < 4; ++dt) {
                const __bf16* vp =
                    vprimeT + ((size_t)(b * HEADS + h)) * 2048u + (dt * 16 + l16) * 32 + 8 * lhalf;
                Frag32B bf;
                bf.q[0] = *(const uint4*)(vp);
                bf.q[1] = *(const uint4*)(vp + 16);
                acc[mt * 8 + hh * 4 + dt] = wmma_bf16(af.v, bf.v, acc[mt * 8 + hh * 4 + dt]);
            }
        }
    }

    #pragma unroll
    for (int mt = 0; mt < 2; ++mt) {
        #pragma unroll
        for (int t = 0; t < 8; ++t) {
            const int col = wn * 128 + t * 16 + l16;
            #pragma unroll
            for (int e = 0; e < 8; ++e) {
                const int row = wm * 32 + mt * 16 + e + 8 * lhalf;
                tileA[row * ASTRIDE + col] = f2bf(acc[mt * 8 + t][e]);
            }
        }
    }
    slab_wait(false, wave);
    __syncthreads();

    // -------------------- GEMM: out = atte @ Wp + bp ------------------------
    #pragma unroll
    for (int t = 0; t < 16; ++t) acc[t] = (v8f){0,0,0,0,0,0,0,0};

    for (int ks = 0; ks < KSTEPS; ++ks) {
        const int buf = ks & 1;
        if (ks + 1 < KSTEPS)
            slab_issue(&wslab[buf ^ 1][0], WpT + (ks + 1) * 32, wave, tid);
        slab_wait(ks + 1 < KSTEPS, wave);
        __syncthreads();

        const __bf16* arow = tileA + ks * 32 + 16 * lhalf;
        Frag32B af0 = load_frag16(arow + (wm * 32 + l16) * ASTRIDE);
        Frag32B af1 = load_frag16(arow + (wm * 32 + 16 + l16) * ASTRIDE);
        Frag32B bcur = load_frag16(&wslab[buf][(wn * 128 + l16) * WSTRIDE + 16 * lhalf]);
        #pragma unroll
        for (int t = 0; t < 8; ++t) {
            Frag32B bnext;
            if (t < 7)
                bnext = load_frag16(&wslab[buf][(wn * 128 + (t + 1) * 16 + l16) * WSTRIDE + 16 * lhalf]);
            acc[t]     = wmma_bf16(af0.v, bcur.v, acc[t]);
            acc[8 + t] = wmma_bf16(af1.v, bcur.v, acc[8 + t]);
            bcur = bnext;
        }
        __syncthreads();
    }

    #pragma unroll
    for (int mt = 0; mt < 2; ++mt) {
        #pragma unroll
        for (int t = 0; t < 8; ++t) {
            const int col = wn * 128 + t * 16 + l16;
            const float bpv = bp[col];
            #pragma unroll
            for (int e = 0; e < 8; ++e) {
                const int row = wm * 32 + mt * 16 + e + 8 * lhalf;
                out[((size_t)(b * N_TOK + n0 + row)) * C + col] = acc[mt * 8 + t][e] + bpv;
            }
        }
    }
}

// ---------------------------------------------------------------------------
extern "C" void kernel_launch(void* const* d_in, const int* in_sizes, int n_in,
                              void* d_out, int out_size, void* d_ws, size_t ws_size,
                              hipStream_t stream) {
    (void)in_sizes; (void)n_in; (void)out_size; (void)ws_size;
    const float* x       = (const float*)d_in[0];
    const float* Wq      = (const float*)d_in[1];
    const float* bq      = (const float*)d_in[2];
    const float* Wk      = (const float*)d_in[3];
    const float* bk      = (const float*)d_in[4];
    const float* Wv      = (const float*)d_in[5];
    const float* bv      = (const float*)d_in[6];
    const float* Wp      = (const float*)d_in[7];
    const float* bp      = (const float*)d_in[8];
    const float* agent_k = (const float*)d_in[9];
    const float* agent_v = (const float*)d_in[10];
    const float* ca_w1   = (const float*)d_in[11];
    const float* ca_w2   = (const float*)d_in[12];
    float* out = (float*)d_out;

    char* ws = (char*)d_ws;
    __bf16*   WqT     = (__bf16*)(ws + WS_WQT);
    __bf16*   WpT     = (__bf16*)(ws + WS_WPT);
    __bf16*   kT      = (__bf16*)(ws + WS_KT);
    float*    vproj   = (float*)(ws + WS_VPROJ);
    float*    gsum    = (float*)(ws + WS_GSUM);
    unsigned* gmax    = (unsigned*)(ws + WS_GMAX);
    __bf16*   vprimeT = (__bf16*)(ws + WS_VPRIME);
    __bf16*   attnA   = (__bf16*)(ws + WS_ATTN);

    // zero attn stats (sum + max, contiguous 4 KB)
    hipMemsetAsync(ws + WS_GSUM, 0, 4096, stream);

    pk_prep_w<<<2048, 256, 0, stream>>>(Wq, Wp, WqT, WpT);
    pk_prep_kv<<<64, 256, 0, stream>>>(agent_k, Wk, bk, agent_v, Wv, bv, kT, vproj);
    pk_qattn<<<B_SZ * 64, 256, 0, stream>>>(x, WqT, bq, kT, attnA, gsum, gmax);
    pk_gate<<<B_SZ, 256, 0, stream>>>(gsum, gmax, ca_w1, ca_w2, vproj, vprimeT);
    pk_avout<<<B_SZ * 64, 256, 0, stream>>>(attnA, vprimeT, WpT, bp, out);
}
